// SqueezedGRU_55765855371921
// MI455X (gfx1250) — compile-verified, matrix operand order checked by reading, not run on hardware
//
#include <hip/hip_runtime.h>
#include <math.h>

// SqueezedGRU for MI455X (gfx1250): bf16 WMMA GEMMs + persistent-grid GRU scan.
// Sizes hardcoded from reference: B=64, T=2000, IN=HID=OUT=512, G=8, L=2.

typedef unsigned short u16;
typedef __attribute__((ext_vector_type(16))) __bf16 v16bf;
typedef __attribute__((ext_vector_type(8)))  float  v8f;

#define DEVINL __device__ __forceinline__

DEVINL u16 f32_to_bf16(float f) {
  __bf16 h = (__bf16)f;
  return __builtin_bit_cast(u16, h);
}

DEVINL v8f zero8() {
  v8f z;
#pragma unroll
  for (int i = 0; i < 8; ++i) z[i] = 0.f;
  return z;
}

// Load a 16x32 bf16 fragment (A layout; B uses same layout on B^T rows).
// gfx1250 wave32 16-bit layout: lanes 0-15 -> K {0..7, 16..23},
// lanes 16-31 -> K {8..15, 24..31}; row = lane & 15.
DEVINL v16bf frag_ld(const u16* base, long ld) {
  const int lane = threadIdx.x & 31;
  const u16* p = base + (long)(lane & 15) * ld + ((lane >> 4) << 3);
  v16bf f;
  float4* h = reinterpret_cast<float4*>(&f);
  h[0] = *reinterpret_cast<const float4*>(p);
  h[1] = *reinterpret_cast<const float4*>(p + 16);
  return f;
}

// Same fragment, sourced from f32 with on-the-fly bf16 convert.
DEVINL v16bf frag_ld_f32(const float* base, long ld) {
  const int lane = threadIdx.x & 31;
  const float* p = base + (long)(lane & 15) * ld + ((lane >> 4) << 3);
  v16bf f;
#pragma unroll
  for (int i = 0; i < 8; ++i) {
    f[i]     = (__bf16)p[i];
    f[i + 8] = (__bf16)p[i + 16];
  }
  return f;
}

DEVINL v8f wmma_bf16(v16bf a, v16bf b, v8f c) {
  return __builtin_amdgcn_wmma_f32_16x16x32_bf16(false, a, false, b, (short)0, c,
                                                 false, false);
}

DEVINL float sigm(float x) { return 1.f / (1.f + __expf(-x)); }

// ---------------------------------------------------------------- weight prep
// gru_k / gru_rk: (2, 512, 1536) f32 -> per-layer transposed bf16 (1536, 512).
__global__ __launch_bounds__(256) void k_conv_gru(const float* __restrict__ Wf,
                                                  u16* __restrict__ Wt) {
  const int l = blockIdx.y;
  const int i = blockIdx.x * 256 + threadIdx.x;
  if (i >= 512 * 1536) return;
  const int k = i / 1536, n = i % 1536;
  Wt[(long)l * 786432 + (long)n * 512 + k] = f32_to_bf16(Wf[(long)l * 786432 + i]);
}

// w_in / w_out: (8, 64, 64) [g][i][o] f32 -> (8, 64, 64) [g][o][i] bf16.
__global__ __launch_bounds__(256) void k_conv_grp(const float* __restrict__ Wf,
                                                  u16* __restrict__ Wt) {
  const int i = blockIdx.x * 256 + threadIdx.x;
  if (i >= 32768) return;
  const int g = i >> 12, r = i & 4095, ii = r >> 6, o = r & 63;
  Wt[g * 4096 + o * 64 + ii] = f32_to_bf16(Wf[i]);
}

// --------------------------------------------- grouped linear + ReLU (f32 in)
// x:(128000,512) f32 -> xin:(128000,512) bf16.  8 waves/block, 1 tile/wave.
__global__ __launch_bounds__(256) void k_gl_in(const float* __restrict__ x,
                                               const u16* __restrict__ wt,
                                               u16* __restrict__ xin) {
  const int w = threadIdx.x >> 5, lane = threadIdx.x & 31;
  const int mt = blockIdx.x;                 // 0..7999
  const int nt = blockIdx.y * 8 + w;         // 0..31
  const int g = nt >> 2;
  const float* abase = x + (long)mt * 16 * 512 + g * 64;
  const u16* bbase = wt + g * 4096 + (long)((nt & 3) * 16) * 64;
  v8f acc = zero8();
#pragma unroll
  for (int kk = 0; kk < 2; ++kk)
    acc = wmma_bf16(frag_ld_f32(abase + kk * 32, 512), frag_ld(bbase + kk * 32, 64), acc);
  const int n = nt * 16 + (lane & 15);
  const int hb = (lane >> 4) << 3;
#pragma unroll
  for (int e = 0; e < 8; ++e) {
    float v = acc[e];
    xin[((long)mt * 16 + hb + e) * 512 + n] = f32_to_bf16(v > 0.f ? v : 0.f);
  }
}

// -------------------------------------------- grouped linear + ReLU (bf16 in)
__global__ __launch_bounds__(256) void k_gl_out(const u16* __restrict__ hseq,
                                                const u16* __restrict__ wt,
                                                float* __restrict__ out) {
  const int w = threadIdx.x >> 5, lane = threadIdx.x & 31;
  const int mt = blockIdx.x;
  const int nt = blockIdx.y * 8 + w;
  const int g = nt >> 2;
  const u16* abase = hseq + (long)mt * 16 * 512 + g * 64;
  const u16* bbase = wt + g * 4096 + (long)((nt & 3) * 16) * 64;
  v8f acc = zero8();
#pragma unroll
  for (int kk = 0; kk < 2; ++kk)
    acc = wmma_bf16(frag_ld(abase + kk * 32, 512), frag_ld(bbase + kk * 32, 64), acc);
  const int n = nt * 16 + (lane & 15);
  const int hb = (lane >> 4) << 3;
#pragma unroll
  for (int e = 0; e < 8; ++e) {
    float v = acc[e];
    // d_out is a 262MB write-once stream: keep it out of L2 (TH=NT).
    __builtin_nontemporal_store(v > 0.f ? v : 0.f,
                                &out[((long)mt * 16 + hb + e) * 512 + n]);
  }
}

// ------------------------------------------------- GRU input projection GEMM
// xproj[m = t*64+b][0..1536) = act(b,t,:) @ K + bias.  act rows are b*2000+t.
// Block: 8 waves arranged 4(M)x2(N); each wave a 32x64 tile (2x4 16x16 tiles,
// 8 WMMAs per 6 fragment loads), K=512.
__global__ __launch_bounds__(256) void k_gemm_xproj(const u16* __restrict__ act,
                                                    const u16* __restrict__ Kt,
                                                    const float* __restrict__ bi,
                                                    float* __restrict__ xp) {
  const int w = threadIdx.x >> 5, lane = threadIdx.x & 31;
  const int wm = w >> 1, wn = w & 1;
  const int mt0 = blockIdx.x * 8 + wm * 2;   // m-tiles (of 8000)
  const int nt0 = blockIdx.y * 8 + wn * 4;   // n-tiles (of 96)
  const u16* abase[2];
#pragma unroll
  for (int mi = 0; mi < 2; ++mi) {
    const int mt = mt0 + mi;
    const int t = mt >> 2;                   // 16-row tile => fixed t, b0..b0+15
    const int b0 = (mt & 3) << 4;
    abase[mi] = act + ((long)b0 * 2000 + t) * 512;
  }
  v8f acc[2][4];
#pragma unroll
  for (int ni = 0; ni < 4; ++ni) {
    const float bv = bi[(nt0 + ni) * 16 + (lane & 15)];   // N fixed per lane
#pragma unroll
    for (int mi = 0; mi < 2; ++mi)
#pragma unroll
      for (int e = 0; e < 8; ++e) acc[mi][ni][e] = bv;
  }
#pragma unroll 2
  for (int kk = 0; kk < 16; ++kk) {
    v16bf a[2], b[4];
#pragma unroll
    for (int mi = 0; mi < 2; ++mi)
      a[mi] = frag_ld(abase[mi] + kk * 32, 1024000);      // row stride = T*512
#pragma unroll
    for (int ni = 0; ni < 4; ++ni)
      b[ni] = frag_ld(Kt + (long)(nt0 + ni) * 8192 + kk * 32, 512);
#pragma unroll
    for (int mi = 0; mi < 2; ++mi)
#pragma unroll
      for (int ni = 0; ni < 4; ++ni)
        acc[mi][ni] = wmma_bf16(a[mi], b[ni], acc[mi][ni]);
  }
  const int hb = (lane >> 4) << 3;
#pragma unroll
  for (int mi = 0; mi < 2; ++mi)
#pragma unroll
    for (int ni = 0; ni < 4; ++ni) {
      const int n = (nt0 + ni) * 16 + (lane & 15);
#pragma unroll
      for (int e = 0; e < 8; ++e)
        // 786MB/layer stream, read exactly once by the scan: bypass L2 (NT)
        // so hseq/xin/weights stay resident in the 192MB L2.
        __builtin_nontemporal_store(
            acc[mi][ni][e], &xp[((long)(mt0 + mi) * 16 + hb + e) * 1536 + n]);
    }
}

// ---------------------------------------------------------------- recurrence
DEVINL void gsync(unsigned* bar, unsigned target) {
  __syncthreads();
  if (threadIdx.x == 0) {
    __hip_atomic_fetch_add(bar, 1u, __ATOMIC_RELEASE, __HIP_MEMORY_SCOPE_AGENT);
    while (__hip_atomic_load(bar, __ATOMIC_ACQUIRE, __HIP_MEMORY_SCOPE_AGENT) < target)
      __builtin_amdgcn_s_sleep(1);
  }
  __syncthreads();
}

// 32 persistent blocks x 128 threads. Block owns h columns [16*bx, 16*bx+16).
// Wave w owns rows b in [16w, 16w+16). RK^T slice staged in LDS (48 KB).
__global__ __launch_bounds__(128) void k_gru_rec(
    const float* __restrict__ xp, const u16* __restrict__ RKt,
    const float* __restrict__ br, u16* __restrict__ hseq,
    float* hf0, float* hf1, u16* hb0, u16* hb1, unsigned* bar) {
  __shared__ u16 ldsB[3 * 16 * 512];         // [gate][col][k], 48 KB
  const int tid = threadIdx.x;
  const int w = tid >> 5, lane = tid & 31;
  const int n0 = blockIdx.x * 16;

  for (int idx = tid; idx < 3 * 16 * 64; idx += 128) {   // 16-byte chunks
    const int row = idx >> 6, ck = idx & 63;
    const int g = row >> 4, c = row & 15;
    const float4 v = *reinterpret_cast<const float4*>(
        RKt + (long)(g * 512 + n0 + c) * 512 + ck * 8);
    *reinterpret_cast<float4*>(&ldsB[row * 512 + ck * 8]) = v;
  }
  for (int idx = tid; idx < 64 * 16; idx += 128) {       // h0 = 0
    const int b = idx >> 4, c = n0 + (idx & 15);
    hf0[b * 512 + c] = 0.f;
    hb0[b * 512 + c] = 0;
  }
  __syncthreads();
  unsigned sy = 1;
  gsync(bar, 32 * sy); ++sy;

  const int j = n0 + (lane & 15);
  const float brz = br[j], brr = br[512 + j], brh = br[1024 + j];
  const int hb = (lane >> 4) << 3;
  float* hcf = hf0; float* hnf = hf1;
  u16* hcb = hb0; u16* hnb = hb1;

  for (int t = 0; t < 2000; ++t) {
    v8f az = zero8(), ar = zero8(), ah = zero8();
    const u16* ab = hcb + (long)(w * 16) * 512;
#pragma unroll 4
    for (int kk = 0; kk < 16; ++kk) {
      v16bf a   = frag_ld(ab + kk * 32, 512);
      v16bf bgz = frag_ld(&ldsB[0 * 8192 + kk * 32], 512);
      v16bf bgr = frag_ld(&ldsB[1 * 8192 + kk * 32], 512);
      v16bf bgh = frag_ld(&ldsB[2 * 8192 + kk * 32], 512);
      az = wmma_bf16(a, bgz, az);
      ar = wmma_bf16(a, bgr, ar);
      ah = wmma_bf16(a, bgh, ah);
    }
#pragma unroll
    for (int e = 0; e < 8; ++e) {
      const int b = w * 16 + hb + e;
      const long xo = ((long)t * 64 + b) * 1536 + j;
      const float xz = __builtin_nontemporal_load(&xp[xo]);          // read-once
      const float xr = __builtin_nontemporal_load(&xp[xo + 512]);    // stream:
      const float xh = __builtin_nontemporal_load(&xp[xo + 1024]);   // TH=NT
      const float z  = sigm(xz + az[e] + brz);
      const float r  = sigm(xr + ar[e] + brr);
      const float hh = tanhf(xh + r * (ah[e] + brh));
      const float hn = z * hcf[b * 512 + j] + (1.f - z) * hh;
      hnf[b * 512 + j] = hn;
      const u16 hnbits = f32_to_bf16(hn);
      hnb[b * 512 + j] = hnbits;
      hseq[((long)b * 2000 + t) * 512 + j] = hnbits;     // re-read next layer: RT
    }
    if (t + 1 < 2000)
      __builtin_prefetch(xp + ((long)(t + 1) * 64 + w * 16) * 1536 + j, 0, 1);
    gsync(bar, 32 * sy); ++sy;                // step boundary + buffer flip
    float* tf = hcf; hcf = hnf; hnf = tf;
    u16* tb = hcb; hcb = hnb; hnb = tb;
  }
}

// ------------------------------------------------------------------- launcher
extern "C" void kernel_launch(void* const* d_in, const int* in_sizes, int n_in,
                              void* d_out, int out_size, void* d_ws, size_t ws_size,
                              hipStream_t stream) {
  (void)in_sizes; (void)n_in; (void)out_size; (void)ws_size;
  const float* x      = (const float*)d_in[0];
  const float* w_in   = (const float*)d_in[1];
  const float* gru_k  = (const float*)d_in[2];
  const float* gru_rk = (const float*)d_in[3];
  const float* gru_b  = (const float*)d_in[4];
  const float* w_out  = (const float*)d_in[5];
  float* out = (float*)d_out;

  char* p = (char*)d_ws;
  auto take = [&](size_t n) { char* q = p; p += (n + 255) & ~(size_t)255; return q; };
  float* xproj = (float*)take((size_t)128000 * 1536 * 4);
  u16*   xin   = (u16*)  take((size_t)128000 * 512 * 2);
  u16*   hseq  = (u16*)  take((size_t)128000 * 512 * 2);
  u16*   Kt    = (u16*)  take((size_t)2 * 1536 * 512 * 2);
  u16*   RKt   = (u16*)  take((size_t)2 * 1536 * 512 * 2);
  u16*   wint  = (u16*)  take((size_t)32768 * 2);
  u16*   woutt = (u16*)  take((size_t)32768 * 2);
  float* hf0   = (float*)take((size_t)64 * 512 * 4);
  float* hf1   = (float*)take((size_t)64 * 512 * 4);
  u16*   hb0   = (u16*)  take((size_t)64 * 512 * 2);
  u16*   hb1   = (u16*)  take((size_t)64 * 512 * 2);
  unsigned* bar = (unsigned*)take(256);

  k_conv_gru<<<dim3(3072, 2), 256, 0, stream>>>(gru_k, Kt);
  k_conv_gru<<<dim3(3072, 2), 256, 0, stream>>>(gru_rk, RKt);
  k_conv_grp<<<128, 256, 0, stream>>>(w_in, wint);
  k_conv_grp<<<128, 256, 0, stream>>>(w_out, woutt);

  k_gl_in<<<dim3(8000, 4), 256, 0, stream>>>(x, wint, xin);

  for (int l = 0; l < 2; ++l) {
    const u16* act = (l == 0) ? xin : hseq;
    const float* bi = gru_b + (long)l * 2 * 1536;
    const float* brp = bi + 1536;
    k_gemm_xproj<<<dim3(1000, 12), 256, 0, stream>>>(act, Kt + (long)l * 786432, bi, xproj);
    hipMemsetAsync(bar, 0, 256, stream);
    k_gru_rec<<<32, 128, 0, stream>>>(xproj, RKt + (long)l * 786432, brp, hseq,
                                      hf0, hf1, hb0, hb1, bar);
  }

  k_gl_out<<<dim3(8000, 4), 256, 0, stream>>>(hseq, woutt, out);
}
